// SimpleLSTM_71012989272503
// MI455X (gfx1250) — compile-verified, matrix operand order checked by reading, not run on hardware
//
#include <hip/hip_runtime.h>

// ---------------- CDNA5 WMMA types ----------------
typedef __attribute__((ext_vector_type(16))) __bf16 bf16x16;
typedef __attribute__((ext_vector_type(8)))  __bf16 bf16x8;
typedef __attribute__((ext_vector_type(8)))  float  f32x8;
typedef __attribute__((ext_vector_type(4)))  int    i32x4;

#define GLOBAL_AS __attribute__((address_space(1)))
#define LDS_AS    __attribute__((address_space(3)))

#if defined(__has_builtin)
#  if __has_builtin(__builtin_amdgcn_global_load_async_to_lds_b128) && \
      __has_builtin(__builtin_amdgcn_s_wait_asynccnt)
#    define USE_ASYNC_LDS 1
#  endif
#endif

__device__ __forceinline__ void copy16_g2l(const void* g, void* l) {
#ifdef USE_ASYNC_LDS
  __builtin_amdgcn_global_load_async_to_lds_b128(
      (GLOBAL_AS i32x4*)(g), (LDS_AS i32x4*)(l), 0, 0);
#else
  *(uint4*)l = *(const uint4*)g;
#endif
}

__device__ __forceinline__ void copy_wait_all() {
#ifdef USE_ASYNC_LDS
  __builtin_amdgcn_s_wait_asynccnt(0);
#endif
}

__device__ __forceinline__ unsigned short f32_to_bf16(float f) {
  unsigned int u = __float_as_uint(f);
  u += 0x7FFFu + ((u >> 16) & 1u);   // round-to-nearest-even
  return (unsigned short)(u >> 16);
}

__device__ __forceinline__ bf16x16 bcat(bf16x8 lo, bf16x8 hi) {
  return __builtin_shufflevector(lo, hi, 0,1,2,3,4,5,6,7,8,9,10,11,12,13,14,15);
}

__device__ __forceinline__ float sigmoidf_fast(float x) {
  return 1.0f / (1.0f + __expf(-x));
}

// ---------------- fp32 -> bf16 conversion (x, straight copy) ----------------
__global__ __launch_bounds__(256) void cvt_f32_bf16_kernel(
    const float* __restrict__ src, unsigned short* __restrict__ dst, long long n) {
  long long i = (long long)blockIdx.x * blockDim.x + threadIdx.x;
  long long stride = (long long)gridDim.x * blockDim.x;
  for (; i < n; i += stride) dst[i] = f32_to_bf16(src[i]);
}

// ---------- fp32 [K][N] -> bf16 [N][K] tiled transpose-convert ----------
__global__ __launch_bounds__(256) void transpose_cvt_kernel(
    const float* __restrict__ src, unsigned short* __restrict__ dst, int K, int N) {
  __shared__ float tile[32][33];
  const int tiles_n = N >> 5;
  const int tk = blockIdx.x / tiles_n;
  const int tn = blockIdx.x % tiles_n;
  const int k0 = tk * 32, n0 = tn * 32;
  const int tx = threadIdx.x & 31, ty = threadIdx.x >> 5;   // 32 x 8
  #pragma unroll
  for (int i = 0; i < 4; ++i)
    tile[ty + 8 * i][tx] = src[(size_t)(k0 + ty + 8 * i) * N + n0 + tx];
  __syncthreads();
  #pragma unroll
  for (int i = 0; i < 4; ++i)
    dst[(size_t)(n0 + ty + 8 * i) * K + k0 + tx] = f32_to_bf16(tile[tx][ty + 8 * i]);
}

// ---------------- state init / finalize ----------------
__global__ __launch_bounds__(256) void init_state_kernel(
    const float* __restrict__ h0, const float* __restrict__ c0,
    unsigned short* __restrict__ hbuf_bf, float* __restrict__ cbuf, int n) {
  int i = blockIdx.x * blockDim.x + threadIdx.x;
  if (i < n) { hbuf_bf[i] = f32_to_bf16(h0[i]); cbuf[i] = c0[i]; }
}

__global__ __launch_bounds__(256) void finalize_kernel(
    const float* __restrict__ out, const float* __restrict__ cfin,
    float* __restrict__ out_c, float* __restrict__ out_h, int H, int T, int n) {
  int i = blockIdx.x * blockDim.x + threadIdx.x;
  if (i < n) {
    int b = i / H, col = i % H;
    out_c[i] = cfin[i];
    out_h[i] = out[((size_t)b * T + (T - 1)) * H + col];
  }
}

// ---------------- Phase 1: xz = x @ Wi + b  (bf16 WMMA, f32 accum) --------
// A:  [M][K] bf16 row-major ; Bt: [N][K] bf16 (pre-transposed Wi)
// C:  stored remapped as [T][Bdim][N]  (row t*Bdim + b)
__global__ __launch_bounds__(128) void gemm_bf16_bias_kernel(
    const unsigned short* __restrict__ A,
    const unsigned short* __restrict__ Bt,
    const float* __restrict__ bias,
    float* __restrict__ C,
    int M, int N, int K, int Tdim, int Bdim) {
  constexpr int TM = 64, TN = 64, TK = 64;
  __shared__ __align__(16) unsigned short lA[2][TM * TK];   // [m][k], double buffered
  __shared__ __align__(16) unsigned short lB[2][TN * TK];   // [n][k]

  const int ntiles = N / TN;
  const int wg_n = blockIdx.x % ntiles;
  const int wg_m = blockIdx.x / ntiles;
  const int m0 = wg_m * TM, n0 = wg_n * TN;
  const int tid = threadIdx.x;
  const int wave = tid >> 5, lane = tid & 31;
  const int l15 = lane & 15, khalf = lane >> 4;

  f32x8 acc[4] = {};

  // staging: 512 uint4 per tile, 4 per thread (row = idx/8, seg = idx%8)
  auto stage = [&](int it, int bufi) {
    const int k0 = it * TK;
    #pragma unroll
    for (int q = 0; q < 4; ++q) {
      const int idx = q * 128 + tid;
      const int row = idx >> 3, seg = idx & 7;
      copy16_g2l(A  + (size_t)(m0 + row) * K + k0 + seg * 8,
                 &lA[bufi][row * TK + seg * 8]);
      copy16_g2l(Bt + (size_t)(n0 + row) * K + k0 + seg * 8,
                 &lB[bufi][row * TK + seg * 8]);
    }
  };

  auto compute = [&](int bufi) {
    const unsigned short* sA = &lA[bufi][0];
    const unsigned short* sB = &lB[bufi][0];
    #pragma unroll
    for (int ks = 0; ks < 2; ++ks) {
      const int ko = ks * 32 + khalf * 8;
      const unsigned short* ap = &sA[(wave * 16 + l15) * TK + ko];
      bf16x16 a  = bcat(*(const bf16x8*)(ap), *(const bf16x8*)(ap + 16));
      // load all four B operands, then issue the four WMMAs back-to-back
      const unsigned short* bp0 = &sB[(0 * 16 + l15) * TK + ko];
      const unsigned short* bp1 = &sB[(1 * 16 + l15) * TK + ko];
      const unsigned short* bp2 = &sB[(2 * 16 + l15) * TK + ko];
      const unsigned short* bp3 = &sB[(3 * 16 + l15) * TK + ko];
      bf16x16 b0 = bcat(*(const bf16x8*)(bp0), *(const bf16x8*)(bp0 + 16));
      bf16x16 b1 = bcat(*(const bf16x8*)(bp1), *(const bf16x8*)(bp1 + 16));
      bf16x16 b2 = bcat(*(const bf16x8*)(bp2), *(const bf16x8*)(bp2 + 16));
      bf16x16 b3 = bcat(*(const bf16x8*)(bp3), *(const bf16x8*)(bp3 + 16));
      acc[0] = __builtin_amdgcn_wmma_f32_16x16x32_bf16(false, a, false, b0, (short)0, acc[0], false, false);
      acc[1] = __builtin_amdgcn_wmma_f32_16x16x32_bf16(false, a, false, b1, (short)0, acc[1], false, false);
      acc[2] = __builtin_amdgcn_wmma_f32_16x16x32_bf16(false, a, false, b2, (short)0, acc[2], false, false);
      acc[3] = __builtin_amdgcn_wmma_f32_16x16x32_bf16(false, a, false, b3, (short)0, acc[3], false, false);
    }
  };

  const int nk = K / TK;
  stage(0, 0);
  copy_wait_all();
  __syncthreads();
  for (int it = 0; it < nk; ++it) {
    if (it + 1 < nk) stage(it + 1, (it + 1) & 1);   // fills other buffer, overlaps compute
    compute(it & 1);
    if (it + 1 < nk) { copy_wait_all(); __syncthreads(); }
  }

  // epilogue: bias + store with [b*T+t] -> [t*B+b] row remap
  #pragma unroll
  for (int j = 0; j < 4; ++j) {
    int n = n0 + j * 16 + l15;
    float bv = bias[n];
    #pragma unroll
    for (int v = 0; v < 8; ++v) {
      int m = m0 + wave * 16 + (khalf ? 8 + v : v);
      int bb = m / Tdim, tt = m % Tdim;
      C[((size_t)tt * Bdim + bb) * N + n] = acc[j][v] + bv;
    }
  }
}

// ---------------- Phase 2: one LSTM timestep (fused 4-gate GEMM + gates) --
// WhT: [4096][1024] bf16 ; xz: [T][64][4096] f32 ; h bf16
__global__ __launch_bounds__(128) void lstm_step_kernel(
    const unsigned short* __restrict__ WhT,
    const float* __restrict__ xz,
    const unsigned short* __restrict__ h_src,
    unsigned short* __restrict__ h_dst,
    float* __restrict__ c_state,
    float* __restrict__ out,
    int t) {
  constexpr int H = 1024, N4 = 4096, Bb = 64, T = 512, TK = 64;
  __shared__ __align__(16) unsigned short lA[2][16 * TK];       // h tile [m][k]
  __shared__ __align__(16) unsigned short lB[2][4 * 16 * TK];   // [g][n][k]
  __shared__ float zb[4 * 256];

  const int jt = blockIdx.x & 63;
  const int mt = blockIdx.x >> 6;
  const int m0 = mt * 16, j0 = jt * 16;
  const int tid = threadIdx.x;
  const int wave = tid >> 5, lane = tid & 31;   // wave == gate index
  const int l15 = lane & 15, khalf = lane >> 4;

  f32x8 acc = {};

  auto stage = [&](int it, int bufi) {
    const int k0 = it * TK;
    // A tile: 128 uint4, 1/thread
    {
      const int row = tid >> 3, seg = tid & 7;
      copy16_g2l(h_src + (size_t)(m0 + row) * H + k0 + seg * 8,
                 &lA[bufi][row * TK + seg * 8]);
    }
    // B tiles: 512 uint4, 4/thread
    #pragma unroll
    for (int q = 0; q < 4; ++q) {
      const int idx = q * 128 + tid;
      const int g = idx >> 7, rem = idx & 127, n = rem >> 3, seg = rem & 7;
      copy16_g2l(WhT + (size_t)(g * H + j0 + n) * H + k0 + seg * 8,
                 &lB[bufi][(g * 16 + n) * TK + seg * 8]);
    }
  };

  auto compute = [&](int bufi) {
    const unsigned short* sA = &lA[bufi][0];
    const unsigned short* sB = &lB[bufi][0];
    const int ko0 = khalf * 8, ko1 = 32 + khalf * 8;
    const unsigned short* ap0 = &sA[l15 * TK + ko0];
    const unsigned short* ap1 = &sA[l15 * TK + ko1];
    const unsigned short* bp0 = &sB[(wave * 16 + l15) * TK + ko0];
    const unsigned short* bp1 = &sB[(wave * 16 + l15) * TK + ko1];
    bf16x16 a0 = bcat(*(const bf16x8*)(ap0), *(const bf16x8*)(ap0 + 16));
    bf16x16 b0 = bcat(*(const bf16x8*)(bp0), *(const bf16x8*)(bp0 + 16));
    bf16x16 a1 = bcat(*(const bf16x8*)(ap1), *(const bf16x8*)(ap1 + 16));
    bf16x16 b1 = bcat(*(const bf16x8*)(bp1), *(const bf16x8*)(bp1 + 16));
    acc = __builtin_amdgcn_wmma_f32_16x16x32_bf16(false, a0, false, b0, (short)0, acc, false, false);
    acc = __builtin_amdgcn_wmma_f32_16x16x32_bf16(false, a1, false, b1, (short)0, acc, false, false);
  };

  const int nk = H / TK;   // 16
  stage(0, 0);
  copy_wait_all();
  __syncthreads();
  for (int it = 0; it < nk; ++it) {
    if (it + 1 < nk) stage(it + 1, (it + 1) & 1);
    compute(it & 1);
    if (it + 1 < nk) { copy_wait_all(); __syncthreads(); }
  }

  // publish gate z tile
  #pragma unroll
  for (int v = 0; v < 8; ++v) {
    int m = khalf ? 8 + v : v;
    zb[wave * 256 + m * 16 + l15] = acc[v];
  }
  __syncthreads();

  // fused elementwise gate math: 256 elems / 128 threads
  for (int e = tid; e < 256; e += 128) {
    int m = e >> 4, n = e & 15;
    int bb = m0 + m;
    int col = j0 + n;
    size_t xzrow = ((size_t)t * Bb + bb) * N4;
    float zi = zb[0 * 256 + e] + xz[xzrow + 0 * H + col];
    float zf = zb[1 * 256 + e] + xz[xzrow + 1 * H + col];
    float zg = zb[2 * 256 + e] + xz[xzrow + 2 * H + col];
    float zo = zb[3 * 256 + e] + xz[xzrow + 3 * H + col];
    size_t sidx = (size_t)bb * H + col;
    float c_old = c_state[sidx];
    float nc = sigmoidf_fast(zf) * c_old + sigmoidf_fast(zi) * tanhf(zg);
    float nh = sigmoidf_fast(zo) * tanhf(nc);
    c_state[sidx] = nc;
    h_dst[sidx] = f32_to_bf16(nh);
    out[((size_t)bb * T + t) * H + col] = nh;
  }
}

// ---------------- host launcher ----------------
extern "C" void kernel_launch(void* const* d_in, const int* in_sizes, int n_in,
                              void* d_out, int out_size, void* d_ws, size_t ws_size,
                              hipStream_t stream) {
  (void)in_sizes; (void)n_in; (void)out_size; (void)ws_size;
  const int B = 64, T = 512, D = 1024, H = 1024, N4 = 4096;
  const long long M = (long long)B * T;               // 32768

  const float* x  = (const float*)d_in[0];
  const float* h0 = (const float*)d_in[1];
  const float* c0 = (const float*)d_in[2];
  const float* Wi = (const float*)d_in[3];
  const float* Wh = (const float*)d_in[4];
  const float* bb = (const float*)d_in[5];

  float* out   = (float*)d_out;                        // [B][T][H]
  float* out_c = out + (size_t)B * T * H;              // c_fin
  float* out_h = out_c + (size_t)B * H;                // h_fin

  char* ws = (char*)d_ws;
  size_t off = 0;
  unsigned short* x_bf   = (unsigned short*)(ws + off); off += (size_t)M * D * 2;    // 64 MB
  unsigned short* WiT_bf = (unsigned short*)(ws + off); off += (size_t)D * N4 * 2;   // 8 MB
  unsigned short* WhT_bf = (unsigned short*)(ws + off); off += (size_t)H * N4 * 2;   // 8 MB
  float* xz = (float*)(ws + off); off += (size_t)M * N4 * 4;                         // 512 MB
  unsigned short* hbuf0 = (unsigned short*)(ws + off); off += (size_t)B * H * 2;
  unsigned short* hbuf1 = (unsigned short*)(ws + off); off += (size_t)B * H * 2;
  float* cstat = (float*)(ws + off); off += (size_t)B * H * 4;

  cvt_f32_bf16_kernel<<<2048, 256, 0, stream>>>(x, x_bf, M * D);
  transpose_cvt_kernel<<<(D / 32) * (N4 / 32), 256, 0, stream>>>(Wi, WiT_bf, D, N4);
  transpose_cvt_kernel<<<(H / 32) * (N4 / 32), 256, 0, stream>>>(Wh, WhT_bf, H, N4);
  init_state_kernel<<<(B * H + 255) / 256, 256, 0, stream>>>(h0, c0, hbuf0, cstat, B * H);

  {
    int grid = (int)((M / 64) * (N4 / 64));            // 512 * 64 = 32768
    gemm_bf16_bias_kernel<<<grid, 128, 0, stream>>>(
        x_bf, WiT_bf, bb, xz, (int)M, N4, D, T, B);
  }

  unsigned short* hsrc = hbuf0;
  unsigned short* hdst = hbuf1;
  for (int t = 0; t < T; ++t) {
    lstm_step_kernel<<<256, 128, 0, stream>>>(WhT_bf, xz, hsrc, hdst, cstat, out, t);
    unsigned short* tmp = hsrc; hsrc = hdst; hdst = tmp;
  }

  finalize_kernel<<<(B * H + 255) / 256, 256, 0, stream>>>(out, cstat, out_c, out_h, H, T, B * H);
}
